// MultiHeadAttention_5935644803251
// MI455X (gfx1250) — matrix operand order repaired and checked by reading, not compiled, not run on hardware
//
#include <hip/hip_runtime.h>
#include <hip/hip_bf16.h>

#define B_      2
#define S_      2048
#define HID_    1024
#define HEADS_  16
#define HEAD_   64

typedef __attribute__((ext_vector_type(16))) __bf16 bf16x16;
typedef __attribute__((ext_vector_type(8)))  float  floatx8;

// ---------------------------------------------------------------------------
// WMMA helper (gfx1250: D = A(16x32 bf16) * B(32x16 bf16) + C(16x16 f32))
// ---------------------------------------------------------------------------
__device__ __forceinline__ floatx8 wmma_bf16(bf16x16 a, bf16x16 b, floatx8 c) {
    return __builtin_amdgcn_wmma_f32_16x16x32_bf16(
        /*neg_a=*/false, a, /*neg_b=*/false, b,
        /*c_mod=*/(short)0, c, /*reuse_a=*/false, /*reuse_b=*/false);
}

// A-fragment (16x32, row-major source, ld in elements).
// Lane m=L%16 holds row m; VGPR0..3 = K (hi*8 + 0..7),
// VGPR4..7 = K (16 + hi*8 + 0..7)  -> two contiguous 16B loads per lane.
__device__ __forceinline__ bf16x16 load_frag_a(const __bf16* base, int ld,
                                               int m, int hi) {
    bf16x16 f;
    const __bf16* p = base + (size_t)m * ld + hi * 8;
    ((uint4*)&f)[0] = *(const uint4*)(p);
    ((uint4*)&f)[1] = *(const uint4*)(p + 16);
    return f;
}

// B-fragment (32x16). B[k,n] read from a row-major matrix R where
// R[n, k] = B[k, n] (weights W for X@W^T; K-matrix for Q@K^T; Vt for P@V).
// Lane n=L%16 holds column n; VGPR0..3 = K (hi*8 + 0..7), VGPR4..7 = K (16 + hi*8 + 0..7).
__device__ __forceinline__ bf16x16 load_frag_b(const __bf16* base, int ld,
                                               int n, int hi) {
    bf16x16 f;
    const __bf16* p = base + (size_t)n * ld + hi * 8;
    ((uint4*)&f)[0] = *(const uint4*)(p);
    ((uint4*)&f)[1] = *(const uint4*)(p + 16);
    return f;
}

// ---------------------------------------------------------------------------
// fp32 -> bf16 conversion
// ---------------------------------------------------------------------------
__global__ void f32_to_bf16_kernel(const float* __restrict__ src,
                                   __bf16* __restrict__ dst, int n) {
    int i = blockIdx.x * blockDim.x + threadIdx.x;
    if (i < n) dst[i] = (__bf16)src[i];
}

// ---------------------------------------------------------------------------
// GEMM: C[M,N] = A[M,K] @ W[N,K]^T + bias[N]
// Block: 256 threads = 8 waves; wave tile 32(M) x 64(N) = 2x4 WMMA subtiles;
// block tile 64 x 256.  8 WMMAs per 12 b128 loads per k-step (~21 FLOP/B).
// ---------------------------------------------------------------------------
template <typename OutT>
__global__ void __launch_bounds__(256)
gemm_bias_kernel(const __bf16* __restrict__ A, const __bf16* __restrict__ W,
                 const float* __restrict__ bias, OutT* __restrict__ C,
                 int M, int N, int K) {
    const int lane = threadIdx.x & 31;
    const int wave = threadIdx.x >> 5;
    const int wm = wave & 1;        // 2 waves along M
    const int wn = wave >> 1;       // 4 waves along N
    const int row0 = blockIdx.x * 64 + wm * 32;
    const int col0 = blockIdx.y * 256 + wn * 64;
    const int m  = lane & 15;
    const int hi = lane >> 4;

    floatx8 acc[2][4] = {};
    for (int k0 = 0; k0 < K; k0 += 32) {
        bf16x16 a0 = load_frag_a(A + (size_t)row0 * K + k0,        K, m, hi);
        bf16x16 a1 = load_frag_a(A + (size_t)(row0 + 16) * K + k0, K, m, hi);
        bf16x16 bf[4];
        #pragma unroll
        for (int tj = 0; tj < 4; ++tj)
            bf[tj] = load_frag_b(W + (size_t)(col0 + tj * 16) * K + k0, K, m, hi);
        #pragma unroll
        for (int tj = 0; tj < 4; ++tj) {
            acc[0][tj] = wmma_bf16(a0, bf[tj], acc[0][tj]);
            acc[1][tj] = wmma_bf16(a1, bf[tj], acc[1][tj]);
        }
    }
    // C layout: lane (m,hi), VGPR r -> element (row0+ti*16 + r + 8*hi, col0+tj*16 + m)
    for (int ti = 0; ti < 2; ++ti)
        for (int tj = 0; tj < 4; ++tj) {
            const int r0 = row0 + ti * 16;
            const int c0 = col0 + tj * 16;
            const float bv = bias[c0 + m];
            #pragma unroll
            for (int r = 0; r < 8; ++r) {
                float v = acc[ti][tj][r] + bv;
                C[(size_t)(r0 + r + 8 * hi) * N + c0 + m] = (OutT)v;
            }
        }
}

// ---------------------------------------------------------------------------
// V transpose: Vp[B*S, HID] -> Vt[B][H][HEAD][S]  (so PV B-frags load contiguous)
// ---------------------------------------------------------------------------
__global__ void transpose_v_kernel(const __bf16* __restrict__ Vp,
                                   __bf16* __restrict__ Vt) {
    int i = blockIdx.x * blockDim.x + threadIdx.x;   // over B*S*HID = 2^22
    if (i >= B_ * S_ * HID_) return;
    int d = i & 63;
    int h = (i >> 6) & (HEADS_ - 1);
    int s = (i >> 10) & (S_ - 1);
    int b = i >> 21;                                  // S_*HID_ = 2^21
    Vt[(((size_t)(b * HEADS_ + h) * HEAD_) + d) * S_ + s] = Vp[i];
}

// ---------------------------------------------------------------------------
// Flash attention: block = 4 waves x 16 query rows = 64 q rows per block.
// KV chunk = 64 keys (16 WMMAs per 64 bpermutes -> shuffle cost halved/key).
// Online softmax; per-QUERY mask (reference: mask[:,None,:,None] hits q axis).
// Qp/Kp: [B*S, HID] bf16 (head slice at h*64).  Vt: [B][H][HEAD][S] bf16.
// Ctx out: [B*S, HID] bf16.
// ---------------------------------------------------------------------------
__global__ void __launch_bounds__(128)
flash_attn_kernel(const __bf16* __restrict__ Qp, const __bf16* __restrict__ Kp,
                  const __bf16* __restrict__ Vt, const int* __restrict__ mask,
                  __bf16* __restrict__ Ctx) {
    __shared__ __align__(16) __bf16 Pl[4][16 * 64];   // per-wave P tile (16x64)

    const int lane = threadIdx.x & 31;
    const int wave = threadIdx.x >> 5;
    const int m  = lane & 15;
    const int hi = lane >> 4;

    const int blk = blockIdx.x;            // b * H * (S/64) blocks
    const int qt = blk & 31;               // S/64 = 32 query tiles
    const int h  = (blk >> 5) & (HEADS_ - 1);
    const int b  = blk >> 9;
    const int q0 = qt * 64 + wave * 16;    // this wave's first query row

    const size_t qbase = ((size_t)b * S_ + q0) * HID_ + h * HEAD_;

    // Q fragments for K-dim (head dim) steps 0 and 32
    bf16x16 qf0 = load_frag_a(Qp + qbase,      HID_, m, hi);
    bf16x16 qf1 = load_frag_a(Qp + qbase + 32, HID_, m, hi);

    // Per-query-row mask (broadcast over keys)
    int mrow[8];
    #pragma unroll
    for (int r = 0; r < 8; ++r) mrow[r] = mask[b * S_ + q0 + r + 8 * hi];

    float mx[8], lsum[8];
    #pragma unroll
    for (int r = 0; r < 8; ++r) { mx[r] = -1e30f; lsum[r] = 0.0f; }
    floatx8 o[4] = {};
    const float scale = 0.125f;            // 1/sqrt(64)

    const __bf16* vhead = Vt + ((size_t)(b * HEADS_ + h) * HEAD_) * S_;

    for (int kv0 = 0; kv0 < S_; kv0 += 64) {
        // ---- S = Q @ K^T for this 16(q) x 64(k) chunk ----
        floatx8 s[4] = {};
        #pragma unroll
        for (int jn = 0; jn < 4; ++jn) {
            const size_t kb = ((size_t)b * S_ + kv0 + jn * 16) * HID_ + h * HEAD_;
            bf16x16 kf0 = load_frag_b(Kp + kb,      HID_, m, hi);
            bf16x16 kf1 = load_frag_b(Kp + kb + 32, HID_, m, hi);
            s[jn] = wmma_bf16(qf0, kf0, s[jn]);
            s[jn] = wmma_bf16(qf1, kf1, s[jn]);
        }
        // ---- scale + query-mask, chunk row max (reduce across 16-lane half) ----
        float nm[8];
        #pragma unroll
        for (int r = 0; r < 8; ++r) {
            float v0 = s[0][r] * scale, v1 = s[1][r] * scale;
            float v2 = s[2][r] * scale, v3 = s[3][r] * scale;
            if (!mrow[r]) { v0 = v1 = v2 = v3 = -1e9f; }
            s[0][r] = v0; s[1][r] = v1; s[2][r] = v2; s[3][r] = v3;
            float v = fmaxf(fmaxf(v0, v1), fmaxf(v2, v3));
            #pragma unroll
            for (int d = 1; d < 16; d <<= 1)
                v = fmaxf(v, __shfl_xor(v, d, 32));
            nm[r] = fmaxf(mx[r], v);
        }
        // ---- P = exp(s - m), row sums, rescale running state, P -> LDS ----
        #pragma unroll
        for (int r = 0; r < 8; ++r) {
            float alpha = __expf(mx[r] - nm[r]);
            mx[r] = nm[r];
            float p0 = __expf(s[0][r] - nm[r]);
            float p1 = __expf(s[1][r] - nm[r]);
            float p2 = __expf(s[2][r] - nm[r]);
            float p3 = __expf(s[3][r] - nm[r]);
            float ps = (p0 + p1) + (p2 + p3);
            #pragma unroll
            for (int d = 1; d < 16; d <<= 1)
                ps += __shfl_xor(ps, d, 32);
            lsum[r] = lsum[r] * alpha + ps;
            #pragma unroll
            for (int j = 0; j < 4; ++j) o[j][r] *= alpha;
            const int row = r + 8 * hi;
            Pl[wave][row * 64 + m]      = (__bf16)p0;
            Pl[wave][row * 64 + 16 + m] = (__bf16)p1;
            Pl[wave][row * 64 + 32 + m] = (__bf16)p2;
            Pl[wave][row * 64 + 48 + m] = (__bf16)p3;
        }
        // ---- re-load P in A-fragment layout (wave-private LDS; no barrier) ----
        bf16x16 pa0 = load_frag_a(&Pl[wave][0],  64, m, hi);
        bf16x16 pa1 = load_frag_a(&Pl[wave][32], 64, m, hi);
        // ---- O += P @ V_chunk  (V chunk is 64 keys x 64 dims) ----
        #pragma unroll
        for (int j = 0; j < 4; ++j) {
            const __bf16* vb_base = vhead + (size_t)(j * 16) * S_ + kv0;
            bf16x16 vb0 = load_frag_b(vb_base,      S_, m, hi);
            bf16x16 vb1 = load_frag_b(vb_base + 32, S_, m, hi);
            o[j] = wmma_bf16(pa0, vb0, o[j]);
            o[j] = wmma_bf16(pa1, vb1, o[j]);
        }
    }
    // ---- normalize and store context (bf16, [B*S, HID] layout) ----
    #pragma unroll
    for (int j = 0; j < 4; ++j) {
        #pragma unroll
        for (int r = 0; r < 8; ++r) {
            float v = o[j][r] / lsum[r];
            const int q = q0 + r + 8 * hi;
            Ctx[((size_t)b * S_ + q) * HID_ + h * HEAD_ + j * 16 + m] = (__bf16)v;
        }
    }
}

// ---------------------------------------------------------------------------
// Host-side orchestration
// ---------------------------------------------------------------------------
extern "C" void kernel_launch(void* const* d_in, const int* in_sizes, int n_in,
                              void* d_out, int out_size, void* d_ws, size_t ws_size,
                              hipStream_t stream) {
    const float* query = (const float*)d_in[0];
    const float* key   = (const float*)d_in[1];
    const float* value = (const float*)d_in[2];
    const int*   maskp = (const int*)d_in[3];
    const float* Wq = (const float*)d_in[4];  const float* bq = (const float*)d_in[5];
    const float* Wk = (const float*)d_in[6];  const float* bk = (const float*)d_in[7];
    const float* Wv = (const float*)d_in[8];  const float* bv = (const float*)d_in[9];
    const float* Wo = (const float*)d_in[10]; const float* bo = (const float*)d_in[11];
    float* out = (float*)d_out;

    const int NT  = B_ * S_;                  // 4096 rows
    const int act = NT * HID_;                // 4,194,304 elements
    const int wel = HID_ * HID_;              // 1,048,576 elements

    __bf16* p   = (__bf16*)d_ws;
    __bf16* xq  = p;  p += act;
    __bf16* xk  = p;  p += act;
    __bf16* xv  = p;  p += act;
    __bf16* wqb = p;  p += wel;
    __bf16* wkb = p;  p += wel;
    __bf16* wvb = p;  p += wel;
    __bf16* wob = p;  p += wel;
    __bf16* Qp  = p;  p += act;
    __bf16* Kp  = p;  p += act;
    __bf16* Vp  = p;  p += act;
    __bf16* Vt  = p;  p += act;
    __bf16* Ctx = p;  p += act;

    // 1) fp32 -> bf16
    f32_to_bf16_kernel<<<(act + 255) / 256, 256, 0, stream>>>(query, xq, act);
    f32_to_bf16_kernel<<<(act + 255) / 256, 256, 0, stream>>>(key,   xk, act);
    f32_to_bf16_kernel<<<(act + 255) / 256, 256, 0, stream>>>(value, xv, act);
    f32_to_bf16_kernel<<<(wel + 255) / 256, 256, 0, stream>>>(Wq, wqb, wel);
    f32_to_bf16_kernel<<<(wel + 255) / 256, 256, 0, stream>>>(Wk, wkb, wel);
    f32_to_bf16_kernel<<<(wel + 255) / 256, 256, 0, stream>>>(Wv, wvb, wel);
    f32_to_bf16_kernel<<<(wel + 255) / 256, 256, 0, stream>>>(Wo, wob, wel);

    // 2) input projections (WMMA GEMMs), bf16 outputs
    dim3 ggrid(NT / 64, HID_ / 256);
    gemm_bias_kernel<__bf16><<<ggrid, 256, 0, stream>>>(xq, wqb, bq, Qp, NT, HID_, HID_);
    gemm_bias_kernel<__bf16><<<ggrid, 256, 0, stream>>>(xk, wkb, bk, Kp, NT, HID_, HID_);
    gemm_bias_kernel<__bf16><<<ggrid, 256, 0, stream>>>(xv, wvb, bv, Vp, NT, HID_, HID_);

    // 3) transpose V per head for contiguous PV fragment loads
    transpose_v_kernel<<<(act + 255) / 256, 256, 0, stream>>>(Vp, Vt);

    // 4) fused attention (QK^T -> online softmax -> PV), WMMA everywhere
    flash_attn_kernel<<<B_ * HEADS_ * (S_ / 64), 128, 0, stream>>>(Qp, Kp, Vt, maskp, Ctx);

    // 5) output projection, fp32 output + bias
    gemm_bias_kernel<float><<<ggrid, 256, 0, stream>>>(Ctx, wob, bo, out, NT, HID_, HID_);
}
